// SelectiveSSMLayer_1090921693692
// MI455X (gfx1250) — matrix operand order repaired and checked by reading, not compiled, not run on hardware
//
#include <hip/hip_runtime.h>
#include <hip/hip_bf16.h>
#include <stdint.h>

// ---------------------------------------------------------------------------
// Problem constants (match reference)
// ---------------------------------------------------------------------------
#define BATCH   16
#define LSEQ    4096
#define DM      256
#define NST     16
#define ROWS    (BATCH * LSEQ)      // 65536 token rows
#define DBC_W   (DM + 2 * NST)      // 288 fused delta|B|C columns
#define CHUNK   64
#define NCHUNK  (LSEQ / CHUNK)      // 64 chunks per sequence

typedef __attribute__((ext_vector_type(16))) __bf16 v16bf;
typedef __attribute__((ext_vector_type(8)))  float  v8f;

struct U4 { unsigned x, y, z, w; };         // 16-byte POD for b128 loads
struct U2 { unsigned x, y; };               // 8-byte POD for b64 stores
union FragU { U4 q[2]; v16bf v; };          // two b128 loads -> one WMMA frag

__device__ __forceinline__ unsigned short f2bf_bits(float f) {
  union { float f; unsigned u; } a; a.f = f;
  return (unsigned short)((a.u + 0x7FFFu + ((a.u >> 16) & 1u)) >> 16);  // RNE
}

__device__ __forceinline__ __bf16 f2bf(float f) {
  union { unsigned short s; __bf16 b; } o; o.s = f2bf_bits(f);
  return o.b;
}

__device__ __forceinline__ float silu(float x) {
  return x / (1.0f + __expf(-x));
}

// ---------------------------------------------------------------------------
// Small prep kernels
// ---------------------------------------------------------------------------
// 4 elements per thread: b128 in, b64 out. n must be a multiple of 4.
__global__ void cvt_bf16x4_kernel(const float* __restrict__ src,
                                  __bf16* __restrict__ dst, int n4) {
  int i = blockIdx.x * blockDim.x + threadIdx.x;
  if (i >= n4) return;
  const U4 v = *reinterpret_cast<const U4*>(src + (size_t)i * 4);
  union { float f; unsigned u; } e0, e1, e2, e3;
  e0.u = v.x; e1.u = v.y; e2.u = v.z; e3.u = v.w;
  U2 o;
  o.x = (unsigned)f2bf_bits(e0.f) | ((unsigned)f2bf_bits(e1.f) << 16);
  o.y = (unsigned)f2bf_bits(e2.f) | ((unsigned)f2bf_bits(e3.f) << 16);
  *reinterpret_cast<U2*>(dst + (size_t)i * 4) = o;
}

__global__ void negexp_kernel(const float* __restrict__ src,
                              float* __restrict__ dst, int n) {
  int i = blockIdx.x * blockDim.x + threadIdx.x;
  if (i < n) dst[i] = -__expf(src[i]);
}

// ---------------------------------------------------------------------------
// bf16 WMMA GEMM:  C[M,N] = A[M,K](bf16) * W[N,K]^T(bf16) + bias, opt softplus
//   - one wave computes a 16 x (NT*16) output strip; NT and K are template
//     params so the K-loop is branch-free and FULLY unrolled -> the scheduler
//     can hoist global_load_b128 fragments ahead of the WMMAs.
//   - A frag: per ISA layout, lane l row = l&15, two 16B runs
//             K in {koff..koff+7} u {koff+16..koff+23}, koff = (l<16)?0:8
//   - B frag: lane l col = l&15, two 16B runs, K in {koff..koff+7} u {koff+16..koff+23}, koff=(l<16)?0:8
// ---------------------------------------------------------------------------
template <int NT, int KDIM>
__global__ __launch_bounds__(256)
void gemm_bf16_kernel(const __bf16* __restrict__ A, const __bf16* __restrict__ W,
                      const float* __restrict__ bias, int biasLen,
                      float* __restrict__ C, int M, int N,
                      int nOffset, int actCols) {
  const unsigned lane = threadIdx.x & 31u;
  const unsigned wave = threadIdx.x >> 5;
  const int mBase  = (blockIdx.x * 8 + (int)wave) * 16;
  const int nGroup = nOffset + blockIdx.y * (NT * 16);
  if (mBase >= M) return;

  const int rowA  = mBase + (int)(lane & 15u);
  const int koffA = (lane < 16u) ? 0 : 8;
  const int koffB = (lane < 16u) ? 0 : 8;
  const int colW  = nGroup + (int)(lane & 15u);

  v8f acc[NT];
  #pragma unroll
  for (int t = 0; t < NT; ++t)
    #pragma unroll
    for (int j = 0; j < 8; ++j) acc[t][j] = 0.0f;

  const __bf16* apBase = A + (size_t)rowA * KDIM + koffA;
  const __bf16* bpBase = W + (size_t)colW * KDIM + koffB;

  #pragma unroll
  for (int kb = 0; kb < KDIM; kb += 32) {
    FragU fa;
    fa.q[0] = *reinterpret_cast<const U4*>(apBase + kb);
    fa.q[1] = *reinterpret_cast<const U4*>(apBase + kb + 16);
    #pragma unroll
    for (int t = 0; t < NT; ++t) {
      FragU fb;
      const __bf16* bp = bpBase + (size_t)t * 16 * KDIM + kb;
      fb.q[0] = *reinterpret_cast<const U4*>(bp);
      fb.q[1] = *reinterpret_cast<const U4*>(bp + 16);
      acc[t] = __builtin_amdgcn_wmma_f32_16x16x32_bf16(
          false, fa.v, false, fb.v, (short)0, acc[t], false, false);
    }
  }

  // Epilogue: lanes 0-15 -> M = j, lanes 16-31 -> M = j + 8
  const int rBase = mBase + ((lane & 16u) ? 8 : 0);
  #pragma unroll
  for (int t = 0; t < NT; ++t) {
    int col = nGroup + t * 16 + (int)(lane & 15u);
    float bv = (col < biasLen) ? bias[col] : 0.0f;
    #pragma unroll
    for (int j = 0; j < 8; ++j) {
      float v = acc[t][j] + bv;
      if (col < actCols)                       // softplus
        v = (v > 20.0f) ? v : log1pf(__expf(v));
      C[(size_t)(rBase + j) * N + col] = v;
    }
  }
}

// ---------------------------------------------------------------------------
// Depthwise conv1d (k=3, pad=1) + SiLU; emits fp32 xc and bf16 copy for GEMM
// ---------------------------------------------------------------------------
__global__ void conv_silu_kernel(const float* __restrict__ xz,
                                 const float* __restrict__ conv_w,
                                 const float* __restrict__ conv_b,
                                 float* __restrict__ xc,
                                 __bf16* __restrict__ xcbf) {
  int idx = blockIdx.x * 256 + threadIdx.x;
  if (idx >= ROWS * DM) return;
  int d   = idx & (DM - 1);
  int row = idx >> 8;                 // b*L + t
  int t   = row & (LSEQ - 1);
  float w0 = conv_w[d * 3 + 0], w1 = conv_w[d * 3 + 1], w2 = conv_w[d * 3 + 2];
  const float* xin = xz + (size_t)row * (2 * DM) + d;   // x_in = cols 0..255 of xz
  float x0 = xin[0];
  float xm = (t > 0)        ? xin[-(2 * DM)] : 0.0f;
  float xp = (t < LSEQ - 1) ? xin[ (2 * DM)] : 0.0f;
  float v = conv_b[d] + w0 * xm + w1 * x0 + w2 * xp;
  v = silu(v);
  xc[idx]   = v;
  xcbf[idx] = f2bf(v);
}

// ---------------------------------------------------------------------------
// Scan phase 1: per (batch, chunk, d) local scan -> P = prod(a), S = local h
// ---------------------------------------------------------------------------
__global__ __launch_bounds__(256)
void scan_part1_kernel(const float* __restrict__ dbc, const float* __restrict__ xc,
                       const float* __restrict__ Aneg,
                       float* __restrict__ Pbuf, float* __restrict__ Sbuf) {
  __shared__ float Bv[CHUNK * NST];
  const int c = blockIdx.x, b = blockIdx.y, d = threadIdx.x;
  const size_t rowBase = (size_t)b * LSEQ + (size_t)c * CHUNK;
  for (int i = threadIdx.x; i < CHUNK * NST; i += 256) {
    int t = i >> 4, n = i & 15;
    Bv[i] = dbc[(rowBase + t) * DBC_W + DM + n];
  }
  __syncthreads();

  float Arow[NST], P[NST], S[NST];
  #pragma unroll
  for (int n = 0; n < NST; ++n) {
    Arow[n] = Aneg[d * NST + n]; P[n] = 1.0f; S[n] = 0.0f;
  }
  for (int t = 0; t < CHUNK; ++t) {
    float dt = dbc[(rowBase + t) * DBC_W + d];
    float du = dt * xc[(rowBase + t) * DM + d];
    #pragma unroll
    for (int n = 0; n < NST; ++n) {
      float a = __expf(dt * Arow[n]);
      P[n] *= a;
      S[n] = a * S[n] + du * Bv[t * NST + n];
    }
  }
  size_t base = ((size_t)(b * NCHUNK + c) * DM + d) * NST;
  #pragma unroll
  for (int n = 0; n < NST; ++n) { Pbuf[base + n] = P[n]; Sbuf[base + n] = S[n]; }
}

// ---------------------------------------------------------------------------
// Scan phase 2: sequential cross-chunk combine, one thread per (b,d,n) state
// ---------------------------------------------------------------------------
__global__ void scan_combine_kernel(const float* __restrict__ Pbuf,
                                    const float* __restrict__ Sbuf,
                                    float* __restrict__ Hinit) {
  int idx = blockIdx.x * 256 + threadIdx.x;   // [0, B*DM*NST) = 65536
  int b = idx >> 12;
  int r = idx & 4095;                          // d*16 + n
  float h = 0.0f;
  for (int c = 0; c < NCHUNK; ++c) {
    size_t e = (((size_t)(b * NCHUNK + c)) << 12) + r;
    Hinit[e] = h;
    h = Pbuf[e] * h + Sbuf[e];
  }
}

// ---------------------------------------------------------------------------
// Scan phase 3: replay with correct h_init; fuse y = (sum_n h*C + u*Dskip)
//               * silu(z); emit bf16 rows for the output GEMM.
// ---------------------------------------------------------------------------
__global__ __launch_bounds__(256)
void scan_part2_kernel(const float* __restrict__ dbc, const float* __restrict__ xc,
                       const float* __restrict__ xz, const float* __restrict__ Aneg,
                       const float* __restrict__ Hinit,
                       const float* __restrict__ Dskip,
                       __bf16* __restrict__ ybf) {
  __shared__ float Bv[CHUNK * NST];
  __shared__ float Cv[CHUNK * NST];
  const int c = blockIdx.x, b = blockIdx.y, d = threadIdx.x;
  const size_t rowBase = (size_t)b * LSEQ + (size_t)c * CHUNK;
  for (int i = threadIdx.x; i < CHUNK * NST; i += 256) {
    int t = i >> 4, n = i & 15;
    Bv[i] = dbc[(rowBase + t) * DBC_W + DM + n];
    Cv[i] = dbc[(rowBase + t) * DBC_W + DM + NST + n];
  }
  __syncthreads();

  float Arow[NST], h[NST];
  size_t hbase = ((size_t)(b * NCHUNK + c) * DM + d) * NST;
  #pragma unroll
  for (int n = 0; n < NST; ++n) {
    Arow[n] = Aneg[d * NST + n];
    h[n] = Hinit[hbase + n];
  }
  float dsk = Dskip[d];
  for (int t = 0; t < CHUNK; ++t) {
    size_t rr = rowBase + t;
    float dt = dbc[rr * DBC_W + d];
    float u  = xc[rr * DM + d];
    float du = dt * u;
    float y  = 0.0f;
    #pragma unroll
    for (int n = 0; n < NST; ++n) {
      float a = __expf(dt * Arow[n]);
      h[n] = a * h[n] + du * Bv[t * NST + n];
      y += h[n] * Cv[t * NST + n];
    }
    float z = xz[rr * (2 * DM) + DM + d];
    ybf[rr * DM + d] = f2bf((y + u * dsk) * silu(z));
  }
}

// ---------------------------------------------------------------------------
// Host launch
// ---------------------------------------------------------------------------
static inline size_t alignUp(size_t v, size_t a) { return (v + a - 1) & ~(a - 1); }

extern "C" void kernel_launch(void* const* d_in, const int* in_sizes, int n_in,
                              void* d_out, int out_size, void* d_ws, size_t ws_size,
                              hipStream_t stream) {
  (void)in_sizes; (void)n_in; (void)out_size; (void)ws_size;
  const float* x       = (const float*)d_in[0];
  const float* A_log   = (const float*)d_in[1];
  const float* D_skip  = (const float*)d_in[2];
  const float* W_B     = (const float*)d_in[3];
  const float* W_C     = (const float*)d_in[4];
  const float* W_delta = (const float*)d_in[5];
  const float* b_delta = (const float*)d_in[6];
  const float* W_in    = (const float*)d_in[7];
  const float* b_in    = (const float*)d_in[8];
  const float* W_out   = (const float*)d_in[9];
  const float* b_out   = (const float*)d_in[10];
  const float* conv_w  = (const float*)d_in[11];
  const float* conv_b  = (const float*)d_in[12];
  float* out = (float*)d_out;

  // Workspace carve (256B aligned)
  char* p = (char*)d_ws;
  size_t off = 0;
  auto carve = [&](size_t bytes) -> char* {
    char* r = p + off;
    off = alignUp(off + bytes, 256);
    return r;
  };
  __bf16* xbf    = (__bf16*)carve((size_t)ROWS * DM * 2);
  __bf16* winbf  = (__bf16*)carve((size_t)(2 * DM) * DM * 2);
  __bf16* wdbcbf = (__bf16*)carve((size_t)DBC_W * DM * 2);
  __bf16* woutbf = (__bf16*)carve((size_t)DM * DM * 2);
  float*  xz     = (float*)carve((size_t)ROWS * 2 * DM * 4);
  float*  xc     = (float*)carve((size_t)ROWS * DM * 4);
  __bf16* xcbf   = (__bf16*)carve((size_t)ROWS * DM * 2);
  float*  dbc    = (float*)carve((size_t)ROWS * DBC_W * 4);
  float*  Aneg   = (float*)carve((size_t)DM * NST * 4);
  float*  Pbuf   = (float*)carve((size_t)BATCH * NCHUNK * DM * NST * 4);
  float*  Sbuf   = (float*)carve((size_t)BATCH * NCHUNK * DM * NST * 4);
  float*  Hinit  = (float*)carve((size_t)BATCH * NCHUNK * DM * NST * 4);
  __bf16* ybf    = (__bf16*)carve((size_t)ROWS * DM * 2);

  auto cvt = [&](const float* s, __bf16* d, int n) {
    int n4 = n / 4;
    cvt_bf16x4_kernel<<<(n4 + 255) / 256, 256, 0, stream>>>(s, d, n4);
  };

  // --- weight / input preparation --------------------------------------
  cvt(x, xbf, ROWS * DM);
  cvt(W_in, winbf, 2 * DM * DM);
  cvt(W_delta, wdbcbf, DM * DM);                       // rows 0..255
  cvt(W_B, wdbcbf + (size_t)DM * DM, NST * DM);        // rows 256..271
  cvt(W_C, wdbcbf + (size_t)(DM + NST) * DM, NST * DM);// rows 272..287
  cvt(W_out, woutbf, DM * DM);
  negexp_kernel<<<(DM * NST + 255) / 256, 256, 0, stream>>>(A_log, Aneg, DM * NST);

  // --- xz = x @ W_in^T + b_in   (WMMA bf16, fp32 acc) ------------------
  gemm_bf16_kernel<4, DM><<<dim3(ROWS / 128, (2 * DM) / 64), 256, 0, stream>>>(
      xbf, winbf, b_in, 2 * DM, xz, ROWS, 2 * DM, /*nOffset=*/0, /*actCols=*/0);

  // --- depthwise conv + SiLU -------------------------------------------
  conv_silu_kernel<<<(ROWS * DM + 255) / 256, 256, 0, stream>>>(
      xz, conv_w, conv_b, xc, xcbf);

  // --- fused [delta | B | C] = xc @ [Wd;WB;WC]^T, softplus on delta ----
  // main 256 cols with NT=4, 32-col tail (B_sel|C_sel) with NT=2
  gemm_bf16_kernel<4, DM><<<dim3(ROWS / 128, DM / 64), 256, 0, stream>>>(
      xcbf, wdbcbf, b_delta, DM, dbc, ROWS, DBC_W, /*nOffset=*/0, /*actCols=*/DM);
  gemm_bf16_kernel<2, DM><<<dim3(ROWS / 128, 1), 256, 0, stream>>>(
      xcbf, wdbcbf, b_delta, DM, dbc, ROWS, DBC_W, /*nOffset=*/DM, /*actCols=*/DM);

  // --- chunked parallel scan (3 phases) --------------------------------
  scan_part1_kernel<<<dim3(NCHUNK, BATCH), 256, 0, stream>>>(dbc, xc, Aneg, Pbuf, Sbuf);
  scan_combine_kernel<<<(BATCH * DM * NST) / 256, 256, 0, stream>>>(Pbuf, Sbuf, Hinit);
  scan_part2_kernel<<<dim3(NCHUNK, BATCH), 256, 0, stream>>>(
      dbc, xc, xz, Aneg, Hinit, D_skip, ybf);

  // --- out = y @ W_out^T + b_out  (WMMA bf16) --------------------------
  gemm_bf16_kernel<4, DM><<<dim3(ROWS / 128, DM / 64), 256, 0, stream>>>(
      ybf, woutbf, b_out, DM, out, ROWS, DM, /*nOffset=*/0, /*actCols=*/0);
}
